// EarlyExitCrossEncoder_1168231104672
// MI455X (gfx1250) — matrix operand order repaired and checked by reading, not compiled, not run on hardware
//
#include <hip/hip_runtime.h>
#include <stdint.h>

// ---------------------------------------------------------------------------
// Types
// ---------------------------------------------------------------------------
typedef __bf16 bf16_t;
typedef __attribute__((ext_vector_type(16))) __bf16       v16bf;
typedef __attribute__((ext_vector_type(8)))  float        v8f;
typedef __attribute__((ext_vector_type(4)))  unsigned int uvec4;
typedef __attribute__((ext_vector_type(8)))  int          ivec8;
typedef __attribute__((ext_vector_type(4)))  int          ivec4;

union Frag  { v16bf v; uvec4 q[2]; };     // 32 bytes = 8 VGPRs (bf16 A/B fragment)
union Pack8 { uvec4 q; bf16_t e[8]; };

// ---------------------------------------------------------------------------
// Problem sizes (match reference)
// ---------------------------------------------------------------------------
constexpr int B_  = 128, S_ = 256, H_ = 384, NH_ = 12, DH_ = 32, F_ = 1536;
constexpr int L_  = 6, NOFF = 5;
constexpr int M_  = B_ * S_;
constexpr float THR_ = 0.1f;
constexpr size_t HH_ = (size_t)H_ * H_;
constexpr size_t HF_ = (size_t)H_ * F_;
constexpr size_t LAYER_W = 4 * HH_ + 2 * HF_;   // bf16 elements of transposed weights / layer

#define DEV static __device__ __forceinline__

// ---------------------------------------------------------------------------
// Tensor Data Mover staging (CDNA5).  Guarded: falls back to cooperative copy.
// ---------------------------------------------------------------------------
#if __has_builtin(__builtin_amdgcn_tensor_load_to_lds) && __has_builtin(__builtin_amdgcn_s_wait_tensorcnt)
#define USE_TDM 1
#else
#define USE_TDM 0
#endif

#if USE_TDM
// DMA a [128 rows x width cols] bf16 tile (row stride = ld elements) into LDS.
// D# layout per CDNA5 ISA (2D tensor, groups 2/3 zero).
DEV void tdm_tile_load(uint32_t lds_off, const bf16_t* gptr, int ld, int rows_total, int width)
{
    uint64_t ga = (uint64_t)(uintptr_t)gptr;
    uvec4 g0;
    g0[0] = 1u;                                   // count=1, is_restore=0, gather off
    g0[1] = lds_off;                              // lds_addr
    g0[2] = (uint32_t)ga;                         // global_addr[31:0]
    g0[3] = (uint32_t)((ga >> 32) & 0x01FFFFFFu)  // global_addr[56:32]
            | 0x80000000u;                        // type=2 ("image") in bits 127:126
    uint32_t td0 = (uint32_t)ld;                  // tensor_dim0 (elements)
    uint32_t td1 = (uint32_t)rows_total;          // tensor_dim1
    ivec8 g1;
    g1[0] = (int)(1u << 16);                                         // data_size=1 (2B)
    g1[1] = (int)((td0 & 0xFFFFu) << 16);                            // tensor_dim0[15:0] @ bit48
    g1[2] = (int)(((td0 >> 16) & 0xFFFFu) | ((td1 & 0xFFFFu) << 16));// dim0 hi | dim1 lo
    g1[3] = (int)(((td1 >> 16) & 0xFFFFu) | ((uint32_t)width << 16));// dim1 hi | tile_dim0
    g1[4] = (int)(128u);                                             // tile_dim1=128, tile_dim2=0
    g1[5] = (int)td0;                                                // tensor_dim0_stride lo32
    g1[6] = 0;                                                       // stride hi | dim1_stride lo
    g1[7] = 0;
    ivec4 z4 = {0, 0, 0, 0};
#if defined(__clang_major__) && (__clang_major__ >= 23)
    ivec8 z8 = {0, 0, 0, 0, 0, 0, 0, 0};
    __builtin_amdgcn_tensor_load_to_lds(g0, g1, z4, z4, z8, 0);
#else
    __builtin_amdgcn_tensor_load_to_lds(g0, g1, z4, z4, 0);
#endif
}
#endif

// ---------------------------------------------------------------------------
// Reductions / math helpers
// ---------------------------------------------------------------------------
DEV float waveReduceSum(float v)
{
    for (int o = 16; o > 0; o >>= 1) v += __shfl_xor(v, o, 32);
    return v;
}

DEV float log_sigmoid(float x)
{
    return (x < 0.f) ? (x - log1pf(expf(x))) : (-log1pf(expf(-x)));
}

// ---------------------------------------------------------------------------
// Weight transpose + convert: WT[n*K + k] = bf16(W[k*N + n])
// ---------------------------------------------------------------------------
__global__ void __launch_bounds__(256) k_transpose_bf16(const float* __restrict__ W,
                                                        bf16_t* __restrict__ WT,
                                                        int K, int N)
{
    int idx = blockIdx.x * 256 + threadIdx.x;
    if (idx >= K * N) return;
    int k = idx / N, n = idx % N;
    WT[(size_t)n * K + k] = (bf16_t)W[idx];
}

// ---------------------------------------------------------------------------
// Embedding + LayerNorm
// ---------------------------------------------------------------------------
__global__ void __launch_bounds__(128) k_embed(const int* __restrict__ ids,
                                               const int* __restrict__ tts,
                                               const float* __restrict__ we,
                                               const float* __restrict__ pe,
                                               const float* __restrict__ te,
                                               const float* __restrict__ g,
                                               const float* __restrict__ bta,
                                               float* __restrict__ h,
                                               bf16_t* __restrict__ hbf)
{
    const int row = blockIdx.x;
    const int s   = row % S_;
    const int tid = threadIdx.x;
    const int id  = ids[row], tt = tts[row];
    float x[3]; float sum = 0.f, sq = 0.f;
#pragma unroll
    for (int i = 0; i < 3; ++i) {
        int c = tid + i * 128;
        float v = we[(size_t)id * H_ + c] + pe[(size_t)s * H_ + c] + te[(size_t)tt * H_ + c];
        x[i] = v; sum += v; sq += v * v;
    }
    __shared__ float s1[4], s2[4], mr[2];
    sum = waveReduceSum(sum); sq = waveReduceSum(sq);
    if ((tid & 31) == 0) { s1[tid >> 5] = sum; s2[tid >> 5] = sq; }
    __syncthreads();
    if (tid == 0) {
        float S1 = s1[0] + s1[1] + s1[2] + s1[3];
        float S2 = s2[0] + s2[1] + s2[2] + s2[3];
        float mean = S1 / H_;
        float var  = S2 / H_ - mean * mean;
        mr[0] = mean; mr[1] = rsqrtf(var + 1e-12f);
    }
    __syncthreads();
    float mean = mr[0], rstd = mr[1];
#pragma unroll
    for (int i = 0; i < 3; ++i) {
        int c = tid + i * 128;
        float y = g[c] * (x[i] - mean) * rstd + bta[c];
        h[(size_t)row * H_ + c]   = y;
        hbf[(size_t)row * H_ + c] = (bf16_t)y;
    }
}

// ---------------------------------------------------------------------------
// Residual add + LayerNorm  (h = LN(h + tmp); also emit bf16 copy)
// ---------------------------------------------------------------------------
__global__ void __launch_bounds__(128) k_add_ln(float* __restrict__ h,
                                                const float* __restrict__ tmp,
                                                const float* __restrict__ g,
                                                const float* __restrict__ bta,
                                                bf16_t* __restrict__ hbf)
{
    const int row = blockIdx.x, tid = threadIdx.x;
    float x[3]; float sum = 0.f, sq = 0.f;
#pragma unroll
    for (int i = 0; i < 3; ++i) {
        int c = tid + i * 128;
        float v = h[(size_t)row * H_ + c] + tmp[(size_t)row * H_ + c];
        x[i] = v; sum += v; sq += v * v;
    }
    __shared__ float s1[4], s2[4], mr[2];
    sum = waveReduceSum(sum); sq = waveReduceSum(sq);
    if ((tid & 31) == 0) { s1[tid >> 5] = sum; s2[tid >> 5] = sq; }
    __syncthreads();
    if (tid == 0) {
        float S1 = s1[0] + s1[1] + s1[2] + s1[3];
        float S2 = s2[0] + s2[1] + s2[2] + s2[3];
        float mean = S1 / H_;
        float var  = S2 / H_ - mean * mean;
        mr[0] = mean; mr[1] = rsqrtf(var + 1e-12f);
    }
    __syncthreads();
    float mean = mr[0], rstd = mr[1];
#pragma unroll
    for (int i = 0; i < 3; ++i) {
        int c = tid + i * 128;
        float y = g[c] * (x[i] - mean) * rstd + bta[c];
        h[(size_t)row * H_ + c]   = y;
        hbf[(size_t)row * H_ + c] = (bf16_t)y;
    }
}

// ---------------------------------------------------------------------------
// Tiled bf16 WMMA GEMM:  C[M,N] = A[M,K] @ BT[N,K]^T + bias
// FUSE: 0 = bias -> bf16   1 = bias+GELU -> bf16   2 = bias -> f32
// Block tile 128x128, 8 waves (4x2), wave tile 32x64.
// K staged in 64-wide tiles, double-buffered via the Tensor Data Mover:
// issue stage k+1, s_wait_tensorcnt(2) to overlap DMA with 16 WMMAs of stage k.
// ---------------------------------------------------------------------------
template <int FUSE>
__global__ void __launch_bounds__(256) k_gemm(const bf16_t* __restrict__ A,
                                              const bf16_t* __restrict__ BT,
                                              const float* __restrict__ bias,
                                              void* __restrict__ Cout,
                                              int M, int N, int K)
{
    __shared__ bf16_t Atile[2][128][64];   // 32 KB
    __shared__ bf16_t Btile[2][128][64];   // 32 KB
    const int tid  = threadIdx.x;
    const int lane = tid & 31, wave = tid >> 5;
    const int wm = wave >> 1, wn = wave & 1;       // 4x2 wave grid
    const int bM = blockIdx.y * 128, bN = blockIdx.x * 128;
    const int nk = K >> 6;                          // number of 64-wide K stages

    v8f acc[2][4];
    const v8f vzero = {0.f, 0.f, 0.f, 0.f, 0.f, 0.f, 0.f, 0.f};
#pragma unroll
    for (int i = 0; i < 2; ++i)
#pragma unroll
        for (int j = 0; j < 4; ++j) acc[i][j] = vzero;

#if USE_TDM
    if (wave == 0) {                                // prologue: stage 0 -> buffer 0
        tdm_tile_load((uint32_t)(uintptr_t)&Atile[0][0][0], A  + (size_t)bM * K, K, M, 64);
        tdm_tile_load((uint32_t)(uintptr_t)&Btile[0][0][0], BT + (size_t)bN * K, K, N, 64);
    }
#endif

    for (int kt = 0; kt < nk; ++kt) {
        const int cur = kt & 1;
        const int k0  = kt << 6;
#if USE_TDM
        if (wave == 0) {
            if (kt + 1 < nk) {                      // issue next stage into other buffer
                const int nb = cur ^ 1;
                tdm_tile_load((uint32_t)(uintptr_t)&Atile[nb][0][0],
                              A  + (size_t)bM * K + k0 + 64, K, M, 64);
                tdm_tile_load((uint32_t)(uintptr_t)&Btile[nb][0][0],
                              BT + (size_t)bN * K + k0 + 64, K, N, 64);
                __builtin_amdgcn_s_wait_tensorcnt(2);   // stage kt complete (in-order)
            } else {
                __builtin_amdgcn_s_wait_tensorcnt(0);
            }
        }
#else
        {
            int r = tid >> 1, c = (tid & 1) << 5;   // 8192 elems / 256 thr = 32 each
#pragma unroll
            for (int j = 0; j < 4; ++j) {
                *(uvec4*)&Atile[cur][r][c + j * 8] =
                    *(const uvec4*)(A  + (size_t)(bM + r) * K + k0 + c + j * 8);
                *(uvec4*)&Btile[cur][r][c + j * 8] =
                    *(const uvec4*)(BT + (size_t)(bN + r) * K + k0 + c + j * 8);
            }
        }
#endif
        __syncthreads();

        if (kt + 2 < nk)                            // pull stage kt+2 toward L2
            __builtin_prefetch(A + (size_t)(bM + (tid >> 1)) * K + k0 + 128, 0, 1);

        const int am  = lane & 15;
        const int akb = (lane < 16) ? 0 : 8;
        const int bkb = (lane < 16) ? 0 : 8;
#pragma unroll
        for (int ks = 0; ks < 64; ks += 32) {
            // A fragments: lane holds row M=l%16, K chunks {kb..kb+7} and {kb+16..kb+23}
            Frag afr[2];
#pragma unroll
            for (int im = 0; im < 2; ++im) {
                const bf16_t* p = &Atile[cur][wm * 32 + im * 16 + am][ks];
                afr[im].q[0] = *(const uvec4*)(p + akb);
                afr[im].q[1] = *(const uvec4*)(p + akb + 16);
            }
            // B fragments from BT (N-major): lane holds col N=l%16, K chunks {kb..kb+7} and {kb+16..kb+23}
#pragma unroll
            for (int in = 0; in < 4; ++in) {
                Frag bfr;
                const bf16_t* p = &Btile[cur][wn * 64 + in * 16 + (lane & 15)][ks];
                bfr.q[0] = *(const uvec4*)(p + bkb);
                bfr.q[1] = *(const uvec4*)(p + bkb + 16);
#pragma unroll
                for (int im = 0; im < 2; ++im)
                    acc[im][in] = __builtin_amdgcn_wmma_f32_16x16x32_bf16(
                        false, afr[im].v, false, bfr.v, (short)0, acc[im][in], false, false);
            }
        }
        __syncthreads();
    }

    // Epilogue: C layout lane l, vgpr r -> row = r + 8*(l/16), col = l%16
#pragma unroll
    for (int im = 0; im < 2; ++im)
#pragma unroll
        for (int in = 0; in < 4; ++in) {
            int col  = bN + wn * 64 + in * 16 + (lane & 15);
            float bv = bias[col];
#pragma unroll
            for (int r = 0; r < 8; ++r) {
                int row = bM + wm * 32 + im * 16 + r + ((lane >> 4) << 3);
                float v = acc[im][in][r] + bv;
                if (FUSE == 1) v = 0.5f * v * (1.0f + erff(v * 0.70710678118f));
                if (FUSE == 2) ((float*)Cout)[(size_t)row * N + col] = v;
                else           ((bf16_t*)Cout)[(size_t)row * N + col] = (bf16_t)v;
            }
        }
}

// ---------------------------------------------------------------------------
// Fused attention: one 128-thread block per (head, batch).
// K rows and V^T staged to LDS ONCE, then 4 waves x 4 q-tiles each:
// scores = Q K^T / sqrt(DH) + maskbias ; softmax in registers ; ctx = P V
// ---------------------------------------------------------------------------
__global__ void __launch_bounds__(128) k_attn(const bf16_t* __restrict__ Q,
                                              const bf16_t* __restrict__ Kt,
                                              const bf16_t* __restrict__ V,
                                              const int* __restrict__ amask,
                                              bf16_t* __restrict__ ctx)
{
    __shared__ bf16_t Ksm[S_][DH_];       // 16 KB : K rows (row-major)
    __shared__ bf16_t VT[DH_][S_];        // 16 KB : V transposed [dh][key]
    __shared__ bf16_t P[4][16][S_];       // 32 KB : per-wave softmax probs
    const int tid  = threadIdx.x;
    const int lane = tid & 31, wave = tid >> 5;
    const int head = blockIdx.x, b = blockIdx.y;
    const size_t hb = (size_t)b * S_;
    const int ho = head * DH_;

    // Cooperative staging of K and V^T (once per (b, head))
    for (int s = tid; s < S_; s += 128) {
        const bf16_t* kp = Kt + (hb + s) * H_ + ho;
        const bf16_t* vp = V  + (hb + s) * H_ + ho;
#pragma unroll
        for (int c = 0; c < 4; ++c) {
            *(uvec4*)&Ksm[s][c * 8] = *(const uvec4*)(kp + c * 8);
            Pack8 pk; pk.q = *(const uvec4*)(vp + c * 8);
#pragma unroll
            for (int e = 0; e < 8; ++e) VT[c * 8 + e][s] = pk.e[e];
        }
    }
    __syncthreads();

    const int m   = lane & 15;
    const int akb = (lane < 16) ? 0 : 8;
    const int bkb = (lane < 16) ? 0 : 8;
    const int rowh = (lane >> 4) << 3;
    const v8f vzero = {0.f, 0.f, 0.f, 0.f, 0.f, 0.f, 0.f, 0.f};

    // mask bias per score column handled by this lane (independent of q tile)
    float mb[16];
#pragma unroll
    for (int t = 0; t < 16; ++t)
        mb[t] = (1.0f - (float)amask[b * S_ + t * 16 + m]) * -1e9f;

    const float scale = 0.1767766953f;              // 1/sqrt(32)

    for (int qt = wave; qt < S_ / 16; qt += 4) {
        // Q fragment (A matrix 16x32, K = DH)
        Frag qf;
        const bf16_t* qp = Q + (hb + qt * 16 + m) * H_ + ho;
        qf.q[0] = *(const uvec4*)(qp + akb);
        qf.q[1] = *(const uvec4*)(qp + akb + 16);

        v8f acc[16];
#pragma unroll
        for (int t = 0; t < 16; ++t) {
            Frag kf;                                // B matrix 32x16 = K^T chunk
            const bf16_t* kp = &Ksm[t * 16 + m][bkb];
            kf.q[0] = *(const uvec4*)(kp);
            kf.q[1] = *(const uvec4*)(kp + 16);
            acc[t] = __builtin_amdgcn_wmma_f32_16x16x32_bf16(
                false, qf.v, false, kf.v, (short)0, vzero, false, false);
        }
#pragma unroll
        for (int t = 0; t < 16; ++t)
#pragma unroll
            for (int r = 0; r < 8; ++r)
                acc[t][r] = acc[t][r] * scale + mb[t];

        // Softmax over 256 columns of each row (16-lane group reduction)
#pragma unroll
        for (int r = 0; r < 8; ++r) {
            float mx = -3.0e38f;
#pragma unroll
            for (int t = 0; t < 16; ++t) mx = fmaxf(mx, acc[t][r]);
            for (int off = 1; off < 16; off <<= 1) mx = fmaxf(mx, __shfl_xor(mx, off, 32));
            float sum = 0.f;
#pragma unroll
            for (int t = 0; t < 16; ++t) { float e = expf(acc[t][r] - mx); acc[t][r] = e; sum += e; }
            for (int off = 1; off < 16; off <<= 1) sum += __shfl_xor(sum, off, 32);
            float inv = 1.0f / sum;
#pragma unroll
            for (int t = 0; t < 16; ++t) acc[t][r] *= inv;
        }

        // Spill P (C layout -> row major LDS) into this wave's private region
#pragma unroll
        for (int t = 0; t < 16; ++t)
#pragma unroll
            for (int r = 0; r < 8; ++r)
                P[wave][rowh + r][t * 16 + m] = (bf16_t)acc[t][r];
        // same-wave LDS ops are in order; DScnt dependency handled by compiler

        // ctx(16x32) = P(16x256) @ V(256x32), K chunks of 32
        v8f cacc[2]; cacc[0] = vzero; cacc[1] = vzero;
#pragma unroll
        for (int kc = 0; kc < 8; ++kc) {
            Frag pf;
            const bf16_t* pp = &P[wave][m][kc * 32 + akb];
            pf.q[0] = *(const uvec4*)(pp);
            pf.q[1] = *(const uvec4*)(pp + 16);
#pragma unroll
            for (int nt = 0; nt < 2; ++nt) {
                Frag vf;
                const bf16_t* vp = &VT[nt * 16 + m][kc * 32 + bkb];
                vf.q[0] = *(const uvec4*)(vp);
                vf.q[1] = *(const uvec4*)(vp + 16);
                cacc[nt] = __builtin_amdgcn_wmma_f32_16x16x32_bf16(
                    false, pf.v, false, vf.v, (short)0, cacc[nt], false, false);
            }
        }
#pragma unroll
        for (int nt = 0; nt < 2; ++nt)
#pragma unroll
            for (int r = 0; r < 8; ++r) {
                int s = qt * 16 + rowh + r;
                ctx[(hb + s) * H_ + ho + nt * 16 + m] = (bf16_t)cacc[nt][r];
            }
    }
}

// ---------------------------------------------------------------------------
// Early-exit off-ramp: logit = h[b,0,:] . Wr[l] + br[l]; entropy gate
// ---------------------------------------------------------------------------
__global__ void __launch_bounds__(128) k_offramp(const float* __restrict__ h,
                                                 const float* __restrict__ Wr,
                                                 const float* __restrict__ br,
                                                 int layer,
                                                 float* scores, int* exitl, float* activef)
{
    const int b = blockIdx.x, tid = threadIdx.x;
    float p = 0.f;
#pragma unroll
    for (int i = 0; i < 3; ++i) {
        int c = tid + i * 128;
        p += h[((size_t)b * S_) * H_ + c] * Wr[layer * H_ + c];
    }
    __shared__ float s1[4];
    p = waveReduceSum(p);
    if ((tid & 31) == 0) s1[tid >> 5] = p;
    __syncthreads();
    if (tid == 0) {
        float logit = s1[0] + s1[1] + s1[2] + s1[3] + br[layer];
        if (activef[b] > 0.5f) {
            float pr  = 1.0f / (1.0f + expf(-logit));
            float ent = -(pr * log_sigmoid(logit) + (1.0f - pr) * log_sigmoid(-logit));
            if (ent < THR_) { scores[b] = logit; exitl[b] = layer; activef[b] = 0.0f; }
        }
    }
}

// Zero rows of exited docs (and refresh bf16 mirror)
__global__ void __launch_bounds__(256) k_mask(float* __restrict__ h,
                                              bf16_t* __restrict__ hbf,
                                              const float* __restrict__ activef)
{
    size_t idx = (size_t)blockIdx.x * 256 + threadIdx.x;
    int b = (int)(idx / ((size_t)S_ * H_));
    float v = h[idx] * activef[b];
    h[idx]   = v;
    hbf[idx] = (bf16_t)v;
}

__global__ void k_init(float* scores, int* exitl, float* activef)
{
    int i = threadIdx.x;
    if (i < B_) { scores[i] = 0.f; exitl[i] = 0; activef[i] = 1.f; }
}

// Pooler + classifier + final select
__global__ void __launch_bounds__(128) k_pooler(const float* __restrict__ h,
                                                const float* __restrict__ Wp,
                                                const float* __restrict__ bp,
                                                const float* __restrict__ Wc,
                                                const float* __restrict__ bc,
                                                const float* __restrict__ scores,
                                                const int* __restrict__ exitl,
                                                const float* __restrict__ activef,
                                                float* __restrict__ out_scores,
                                                int* __restrict__ out_exit,
                                                int* __restrict__ exit_final)
{
    const int b = blockIdx.x, tid = threadIdx.x;
    __shared__ float pooled[H_];
    const float* hr = h + (size_t)b * S_ * H_;   // row 0 of doc b
    for (int j = tid; j < H_; j += 128) {
        float acc = bp[j];
        for (int k = 0; k < H_; ++k) acc += hr[k] * Wp[(size_t)k * H_ + j];
        pooled[j] = tanhf(acc);
    }
    __syncthreads();
    float p = 0.f;
#pragma unroll
    for (int i = 0; i < 3; ++i) { int c = tid + i * 128; p += pooled[c] * Wc[c]; }
    __shared__ float s1[4];
    p = waveReduceSum(p);
    if ((tid & 31) == 0) s1[tid >> 5] = p;
    __syncthreads();
    if (tid == 0) {
        float logit = s1[0] + s1[1] + s1[2] + s1[3] + bc[0];
        bool act = activef[b] > 0.5f;
        out_scores[b] = act ? logit : scores[b];
        int el = act ? NOFF : exitl[b];
        out_exit[b]   = el;
        exit_final[b] = el;
    }
}

__global__ void k_counts(const int* __restrict__ exit_final, int* __restrict__ out_counts)
{
    __shared__ int cnt[NOFF + 1];
    int tid = threadIdx.x;
    if (tid < NOFF + 1) cnt[tid] = 0;
    __syncthreads();
    for (int i = tid; i < B_; i += blockDim.x) atomicAdd(&cnt[exit_final[i]], 1);
    __syncthreads();
    if (tid < NOFF + 1) out_counts[tid] = cnt[tid];
}

// ---------------------------------------------------------------------------
// Host orchestration
// ---------------------------------------------------------------------------
extern "C" void kernel_launch(void* const* d_in, const int* in_sizes, int n_in,
                              void* d_out, int out_size, void* d_ws, size_t ws_size,
                              hipStream_t stream)
{
    const int*   ids = (const int*)d_in[0];
    const int*   am  = (const int*)d_in[1];
    const int*   tt  = (const int*)d_in[2];
    const float* we  = (const float*)d_in[3];
    const float* pe  = (const float*)d_in[4];
    const float* te  = (const float*)d_in[5];
    const float* eg  = (const float*)d_in[6];
    const float* eb  = (const float*)d_in[7];
    const float* Wq  = (const float*)d_in[8];
    const float* bq  = (const float*)d_in[9];
    const float* Wk  = (const float*)d_in[10];
    const float* bk  = (const float*)d_in[11];
    const float* Wv  = (const float*)d_in[12];
    const float* bv  = (const float*)d_in[13];
    const float* Wo  = (const float*)d_in[14];
    const float* bo  = (const float*)d_in[15];
    const float* l1g = (const float*)d_in[16];
    const float* l1b = (const float*)d_in[17];
    const float* Wi  = (const float*)d_in[18];
    const float* bi  = (const float*)d_in[19];
    const float* Wfo = (const float*)d_in[20];
    const float* bfo = (const float*)d_in[21];
    const float* l2g = (const float*)d_in[22];
    const float* l2b = (const float*)d_in[23];
    const float* Wp  = (const float*)d_in[24];
    const float* bp  = (const float*)d_in[25];
    const float* Wc  = (const float*)d_in[26];
    const float* bc  = (const float*)d_in[27];
    const float* Wr  = (const float*)d_in[28];
    const float* br  = (const float*)d_in[29];

    // ---- carve workspace ----
    char*  wsb = (char*)d_ws;
    size_t off = 0;
    auto alloc = [&](size_t bytes) {
        void* p = wsb + off;
        off += (bytes + 255) & ~(size_t)255;
        return p;
    };
    float*  h   = (float*)alloc((size_t)M_ * H_ * 4);
    float*  tmp = (float*)alloc((size_t)M_ * H_ * 4);
    bf16_t* hbf = (bf16_t*)alloc((size_t)M_ * H_ * 2);
    bf16_t* qb  = (bf16_t*)alloc((size_t)M_ * H_ * 2);
    bf16_t* kb  = (bf16_t*)alloc((size_t)M_ * H_ * 2);
    bf16_t* vb  = (bf16_t*)alloc((size_t)M_ * H_ * 2);
    bf16_t* cb  = (bf16_t*)alloc((size_t)M_ * H_ * 2);
    bf16_t* fb  = (bf16_t*)alloc((size_t)M_ * F_ * 2);
    bf16_t* wT  = (bf16_t*)alloc((size_t)L_ * LAYER_W * 2);
    float*  st_scores = (float*)alloc(B_ * 4);
    int*    st_exit   = (int*)alloc(B_ * 4);
    float*  st_act    = (float*)alloc(B_ * 4);
    int*    st_exitf  = (int*)alloc(B_ * 4);

    // ---- transpose+convert weights once per launch ----
    for (int l = 0; l < L_; ++l) {
        bf16_t* base = wT + (size_t)l * LAYER_W;
        auto tr = [&](const float* src, bf16_t* dst, int K, int N) {
            int n = K * N;
            k_transpose_bf16<<<(n + 255) / 256, 256, 0, stream>>>(src, dst, K, N);
        };
        tr(Wq  + (size_t)l * HH_, base,            H_, H_);
        tr(Wk  + (size_t)l * HH_, base + HH_,      H_, H_);
        tr(Wv  + (size_t)l * HH_, base + 2 * HH_,  H_, H_);
        tr(Wo  + (size_t)l * HH_, base + 3 * HH_,  H_, H_);
        tr(Wi  + (size_t)l * HF_, base + 4 * HH_,  H_, F_);
        tr(Wfo + (size_t)l * HF_, base + 4 * HH_ + HF_, F_, H_);
    }

    k_init<<<1, 128, 0, stream>>>(st_scores, st_exit, st_act);
    k_embed<<<M_, 128, 0, stream>>>(ids, tt, we, pe, te, eg, eb, h, hbf);

    const dim3 g384(H_ / 128, M_ / 128);
    const dim3 g1536(F_ / 128, M_ / 128);
    const int maskBlocks = (int)(((size_t)M_ * H_) / 256);

    for (int l = 0; l < L_; ++l) {
        bf16_t* base = wT + (size_t)l * LAYER_W;
        // QKV projections (bf16 out)
        k_gemm<0><<<g384, 256, 0, stream>>>(hbf, base,           bq + (size_t)l * H_, qb, M_, H_, H_);
        k_gemm<0><<<g384, 256, 0, stream>>>(hbf, base + HH_,     bk + (size_t)l * H_, kb, M_, H_, H_);
        k_gemm<0><<<g384, 256, 0, stream>>>(hbf, base + 2 * HH_, bv + (size_t)l * H_, vb, M_, H_, H_);
        // fused attention (one block per (head, batch))
        k_attn<<<dim3(NH_, B_), 128, 0, stream>>>(qb, kb, vb, am, cb);
        // output projection (f32) + residual LN
        k_gemm<2><<<g384, 256, 0, stream>>>(cb, base + 3 * HH_, bo + (size_t)l * H_, tmp, M_, H_, H_);
        k_add_ln<<<M_, 128, 0, stream>>>(h, tmp, l1g + (size_t)l * H_, l1b + (size_t)l * H_, hbf);
        // FFN: GEMM+GELU (bf16) then GEMM (f32) + residual LN
        k_gemm<1><<<g1536, 256, 0, stream>>>(hbf, base + 4 * HH_, bi + (size_t)l * F_, fb, M_, F_, H_);
        k_gemm<2><<<g384, 256, 0, stream>>>(fb, base + 4 * HH_ + HF_, bfo + (size_t)l * H_, tmp, M_, H_, F_);
        k_add_ln<<<M_, 128, 0, stream>>>(h, tmp, l2g + (size_t)l * H_, l2b + (size_t)l * H_, hbf);
        // early-exit off-ramp
        if (l < NOFF) {
            k_offramp<<<B_, 128, 0, stream>>>(h, Wr, br, l, st_scores, st_exit, st_act);
            k_mask<<<maskBlocks, 256, 0, stream>>>(h, hbf, st_act);
        }
    }

    float* out_scores = (float*)d_out;
    int*   out_exit   = (int*)((float*)d_out + B_);
    int*   out_counts = (int*)((float*)d_out + 2 * B_);
    k_pooler<<<B_, 128, 0, stream>>>(h, Wp, bp, Wc, bc, st_scores, st_exit, st_act,
                                     out_scores, out_exit, st_exitf);
    k_counts<<<1, 128, 0, stream>>>(st_exitf, out_counts);

    (void)in_sizes; (void)n_in; (void)out_size; (void)ws_size;
}
